// SS2D_31258771980443
// MI455X (gfx1250) — matrix operand order repaired and checked by reading, not compiled, not run on hardware
//
#include <hip/hip_runtime.h>
#include <hip/hip_bf16.h>

// ---------------- problem constants ----------------
#define Bn 2
#define Hh 56
#define Ww 56
#define Cc 192
#define Ll (Hh * Ww)          // 3136
#define DI 384                // D_INNER
#define DS 8                  // D_STATE
#define DR 12                 // DT_RANK
#define ND 4                  // directions
#define XDBL 32               // padded 28 -> 32

typedef __attribute__((ext_vector_type(16))) __bf16        v16bf;
typedef __attribute__((ext_vector_type(8)))  float         v8f;
typedef __attribute__((ext_vector_type(8)))  unsigned int  v8u;

// ---------------- helpers ----------------
__device__ __forceinline__ unsigned short f2bf(float f) {
    unsigned int u = __float_as_uint(f);
    u += 0x7FFFu + ((u >> 16) & 1u);       // round-to-nearest-even
    return (unsigned short)(u >> 16);
}
__device__ __forceinline__ float silu(float v) {
    return v / (1.f + __expf(-v));
}
__device__ __forceinline__ int seq_to_spatial(int d, int l) {
    switch (d) {
        case 0: return l;
        case 1: return Ll - 1 - l;
        case 2: { int w = l / Hh, h = l % Hh; return h * Ww + w; }
        default:{ int lr = Ll - 1 - l; int w = lr / Hh, h = lr % Hh; return h * Ww + w; }
    }
}
// pack two 16-byte loads into a 32-byte bf16 fragment
__device__ __forceinline__ v16bf pack_frag(const unsigned short* p0,
                                           const unsigned short* p1) {
    uint4 a = *(const uint4*)p0;
    uint4 b = *(const uint4*)p1;
    v8u u;
    u[0] = a.x; u[1] = a.y; u[2] = a.z; u[3] = a.w;
    u[4] = b.x; u[5] = b.y; u[6] = b.z; u[7] = b.w;
    return __builtin_bit_cast(v16bf, u);
}

// ---------------- WMMA GEMM: one wave = 16 x (16*NT4) f32 tile ----------------
// A: row-major [M][K] bf16.  B: TRANSPOSED, [N][K] bf16 (contiguous per column).
// MODE 0: plain f32 store
// MODE 1: split store: col<384 -> C0 (x_in), else -> C1 (z), both stride ldc
// MODE 2: +bias, SiLU, store f32 (C0) and bf16 (Cbf)
// MODE 5: +bias (shared across batch), store f32
template <int MODE, int NT4>
__global__ __launch_bounds__(256) void wmma_gemm_kernel(
    const unsigned short* __restrict__ A, long sA, int lda,
    const unsigned short* __restrict__ Bt, long sB,
    int Mt, int Ng, int K, int nbat,
    float* __restrict__ C0, long sC0, int ldc,
    float* __restrict__ C1,
    unsigned short* __restrict__ Cbf,
    const float* __restrict__ bias, int sBias)
{
    int wave  = blockIdx.x * (blockDim.x >> 5) + (threadIdx.x >> 5);
    int total = Mt * Ng * nbat;
    if (wave >= total) return;
    int tg = wave % Ng;
    int t2 = wave / Ng;
    int tm = t2 % Mt;
    int bt = t2 / Mt;

    int lane = threadIdx.x & 31;
    int half = lane >> 4;
    int l16  = lane & 15;
    int h8   = half * 8;
    int h16  = half * 16;

    // A fragment rows: lanes 0-15 and 16-31 both carry M = l16; half selects K split
    const unsigned short* Arow = A + (long)bt * sA + (long)(tm * 16 + l16) * lda;

    const unsigned short* Bcol[NT4];
    int coln[NT4];
#pragma unroll
    for (int j = 0; j < NT4; ++j) {
        coln[j] = (tg * NT4 + j) * 16 + l16;
        Bcol[j] = Bt + (long)bt * sB + (long)coln[j] * K;
    }

    v8f acc[NT4];
#pragma unroll
    for (int j = 0; j < NT4; ++j) acc[j] = {};

    for (int kk = 0; kk < K; kk += 32) {
        // A layout: lanes 0-15 hold K kk..kk+7 & kk+16..kk+23; lanes 16-31 shifted by 8
        v16bf av = pack_frag(Arow + kk + h8, Arow + kk + h8 + 16);
#pragma unroll
        for (int j = 0; j < NT4; ++j) {
            // B layout: lanes 0-15 hold K kk..kk+7 & kk+16..kk+23; lanes 16-31 shifted by 8
            v16bf bv = pack_frag(Bcol[j] + kk + h8, Bcol[j] + kk + h8 + 16);
            acc[j] = __builtin_amdgcn_wmma_f32_16x16x32_bf16(
                false, av, false, bv, (short)0, acc[j], false, false);
        }
    }

#pragma unroll
    for (int j = 0; j < NT4; ++j) {
        int col = coln[j];
#pragma unroll
        for (int r = 0; r < 8; ++r) {
            int row = tm * 16 + h8 + r;   // C/D: VGPR r -> M = r + half*8, N = l16
            float v = acc[j][r];
            if (MODE == 1) {
                long base = (long)bt * sC0 + (long)row * ldc;
                if (col < DI) C0[base + col] = v;
                else          C1[base + (col - DI)] = v;
            } else if (MODE == 2) {
                v += bias[bt * sBias + col];
                float s = silu(v);
                long idx = (long)bt * sC0 + (long)row * ldc + col;
                C0[idx]  = s;
                Cbf[idx] = f2bf(s);
            } else if (MODE == 5) {
                v += bias[col];
                C0[(long)bt * sC0 + (long)row * ldc + col] = v;
            } else {
                C0[(long)bt * sC0 + (long)row * ldc + col] = v;
            }
        }
    }
}

// ---------------- weight conversion: f32 [K][N] -> bf16 TRANSPOSED [Npad][K] ----
__global__ void cvt_t_kernel(const float* __restrict__ s,
                             unsigned short* __restrict__ d,
                             int K, int N, int Npad, long n) {
    long i = blockIdx.x * (long)blockDim.x + threadIdx.x;
    if (i >= n) return;
    int k  = (int)(i % K);
    int nn = (int)((i / K) % Npad);
    int b  = (int)(i / ((long)K * Npad));
    float v = (nn < N) ? s[((long)b * K + k) * N + nn] : 0.f;
    d[i] = f2bf(v);
}

// gather the 4 scan orderings of x and convert to bf16
__global__ void gather_seq_kernel(const float* __restrict__ x,
                                  unsigned short* __restrict__ seq, int n) {
    int i = blockIdx.x * blockDim.x + threadIdx.x;
    if (i >= n) return;
    int c = i % Cc;
    int l = (i / Cc) % Ll;
    int b = (i / (Cc * Ll)) % Bn;
    int d = i / (Cc * Ll * Bn);
    int s = seq_to_spatial(d, l);
    seq[i] = f2bf(x[((long)b * Ll + s) * Cc + c]);
}

// depthwise causal conv, width 4, per channel
__global__ void dwconv_kernel(const float* __restrict__ xin,
                              const float* __restrict__ dwk,
                              unsigned short* __restrict__ out, int n) {
    int i = blockIdx.x * blockDim.x + threadIdx.x;
    if (i >= n) return;
    int c  = i % DI;
    int l  = (i / DI) % Ll;
    int db = i / (DI * Ll);           // d*Bn + b
    int d  = db / Bn;
    float acc = 0.f;
#pragma unroll
    for (int j = 0; j < 4; ++j) {
        int ls = l - 3 + j;
        if (ls >= 0)
            acc += dwk[(d * 4 + j) * DI + c] * xin[((long)db * Ll + ls) * DI + c];
    }
    out[i] = f2bf(acc);
}

// delta = softplus(x_dbl[:, :12] @ W_dt + b_dt)
__global__ void delta_kernel(const float* __restrict__ xdbl,
                             const float* __restrict__ wdt,
                             const float* __restrict__ bdt,
                             float* __restrict__ delta, int n) {
    int i = blockIdx.x * blockDim.x + threadIdx.x;
    if (i >= n) return;
    int  c   = i % DI;
    long row = i / DI;                 // (d*Bn+b)*Ll + l
    int  d   = (int)(row / ((long)Bn * Ll));
    const float* xr = xdbl + row * XDBL;
    float acc = bdt[d * DI + c];
#pragma unroll
    for (int r = 0; r < DR; ++r)
        acc += xr[r] * wdt[(d * DR + r) * DI + c];
    delta[i] = (acc > 20.f) ? acc : log1pf(__expf(acc));
}

// sequential selective scan: one lane per (dir, b, channel); state in VGPRs;
// shared B/C rows staged through LDS (16 loads instead of 384 per step)
__global__ __launch_bounds__(DI) void scan_kernel(
    const float* __restrict__ delta, const float* __restrict__ xconv,
    const float* __restrict__ xdbl,  const float* __restrict__ zbuf,
    const float* __restrict__ Alog,  const float* __restrict__ Dpar,
    unsigned short* __restrict__ ybf)
{
    __shared__ float sbc[2 * DS];      // [0..7]=B row, [8..15]=C row
    int db = blockIdx.x;               // d*Bn + b
    int d  = db / Bn;
    int t  = threadIdx.x;

    float Ar[DS], h[DS];
#pragma unroll
    for (int nn = 0; nn < DS; ++nn) {
        Ar[nn] = -__expf(Alog[((long)d * DI + t) * DS + nn]);
        h[nn]  = 0.f;
    }
    float Dp = Dpar[d * DI + t];

    long rowbase = (long)db * Ll;
    for (int l = 0; l < Ll; ++l) {
        long row = rowbase + l;
        if (l + 8 < Ll) {   // keep the recurrence fed: global_prefetch_b8
            __builtin_prefetch(&delta[(row + 8) * DI + t], 0, 0);
            __builtin_prefetch(&xconv[(row + 8) * DI + t], 0, 0);
        }
        if (t < 2 * DS) sbc[t] = xdbl[row * XDBL + DR + t];
        __syncthreads();

        float dv = delta[row * DI + t];
        float xc = xconv[row * DI + t];
        float y = 0.f;
#pragma unroll
        for (int nn = 0; nn < DS; ++nn) {
            float dA = __expf(dv * Ar[nn]);
            h[nn] = dA * h[nn] + dv * sbc[nn] * xc;
            y += sbc[DS + nn] * h[nn];
        }
        float zv = zbuf[row * DI + t];
        y = (y + xc * Dp) * silu(zv);
        ybf[row * DI + t] = f2bf(y);
        __syncthreads();
    }
}

// un-permute + sum 4 directions, convert to bf16 for the final GEMM
__global__ void merge_kernel(const float* __restrict__ om,
                             unsigned short* __restrict__ mg, int n) {
    int i = blockIdx.x * blockDim.x + threadIdx.x;
    if (i >= n) return;
    int c = i % Cc;
    int s = (i / Cc) % Ll;
    int b = i / (Cc * Ll);
    int hh = s / Ww, ww = s % Ww;
    int l2 = ww * Hh + hh;
    long stride = (long)Bn * Ll * Cc;
    long rb = (long)b * Ll * Cc;
    float v = om[0 * stride + rb + (long)s            * Cc + c]
            + om[1 * stride + rb + (long)(Ll - 1 - s) * Cc + c]
            + om[2 * stride + rb + (long)l2           * Cc + c]
            + om[3 * stride + rb + (long)(Ll - 1 - l2)* Cc + c];
    mg[i] = f2bf(v);
}

// ---------------- host launcher ----------------
extern "C" void kernel_launch(void* const* d_in, const int* in_sizes, int n_in,
                              void* d_out, int out_size, void* d_ws, size_t ws_size,
                              hipStream_t stream) {
    (void)in_sizes; (void)n_in; (void)out_size; (void)ws_size;

    const float* x      = (const float*)d_in[0];
    const float* W_in   = (const float*)d_in[1];
    const float* dwk    = (const float*)d_in[2];
    const float* pwk    = (const float*)d_in[3];
    const float* cbias  = (const float*)d_in[4];
    const float* W_x    = (const float*)d_in[5];
    const float* W_dt   = (const float*)d_in[6];
    const float* b_dt   = (const float*)d_in[7];
    const float* A_log  = (const float*)d_in[8];
    const float* D_par  = (const float*)d_in[9];
    const float* W_outm = (const float*)d_in[10];
    const float* W_out  = (const float*)d_in[11];
    const float* b_out  = (const float*)d_in[12];
    float* out = (float*)d_out;

    // workspace carve-out
    char* p = (char*)d_ws;
    auto alloc = [&](size_t bytes) -> void* {
        void* r = (void*)p;
        p += (bytes + 255) & ~(size_t)255;
        return r;
    };
    const long nBL = (long)Bn * Ll;            // 6272 rows per direction
    unsigned short* seq_bf    = (unsigned short*)alloc((size_t)ND * nBL * Cc * 2);
    unsigned short* win_t     = (unsigned short*)alloc((size_t)ND * 2 * DI * Cc * 2);  // [768][192]
    unsigned short* pwk_t     = (unsigned short*)alloc((size_t)ND * DI * DI * 2);      // [384][384]
    unsigned short* wx_t      = (unsigned short*)alloc((size_t)ND * XDBL * DI * 2);    // [32][384]
    unsigned short* woutm_t   = (unsigned short*)alloc((size_t)ND * Cc * DI * 2);      // [192][384]
    unsigned short* wout_t    = (unsigned short*)alloc((size_t)Cc * Cc * 2);           // [192][192]
    float*          x_in      = (float*)alloc((size_t)ND * nBL * DI * 4);
    float*          z_buf     = (float*)alloc((size_t)ND * nBL * DI * 4);
    unsigned short* conv_bf   = (unsigned short*)alloc((size_t)ND * nBL * DI * 2);
    float*          x_conv    = (float*)alloc((size_t)ND * nBL * DI * 4);
    unsigned short* x_conv_bf = (unsigned short*)alloc((size_t)ND * nBL * DI * 2);
    float*          x_dbl     = (float*)alloc((size_t)ND * nBL * XDBL * 4);
    float*          delta     = (float*)alloc((size_t)ND * nBL * DI * 4);
    unsigned short* y_bf      = (unsigned short*)alloc((size_t)ND * nBL * DI * 2);
    float*          outm      = (float*)alloc((size_t)ND * nBL * Cc * 4);
    unsigned short* merged_bf = (unsigned short*)alloc((size_t)nBL * Cc * 2);

    const int T = 256;
    auto nb = [](long n, int t) { return (unsigned)((n + t - 1) / t); };

    // 1) weight conversions (bf16, transposed to [N][K] for contiguous fragments)
    cvt_t_kernel<<<nb((long)ND * 2 * DI * Cc, T), T, 0, stream>>>(W_in,   win_t,   Cc, 2 * DI, 2 * DI, (long)ND * 2 * DI * Cc);
    cvt_t_kernel<<<nb((long)ND * DI * DI, T), T, 0, stream>>>(pwk,    pwk_t,   DI, DI, DI,          (long)ND * DI * DI);
    cvt_t_kernel<<<nb((long)ND * XDBL * DI, T), T, 0, stream>>>(W_x,    wx_t,    DI, DR + 2 * DS, XDBL, (long)ND * XDBL * DI);
    cvt_t_kernel<<<nb((long)ND * Cc * DI, T), T, 0, stream>>>(W_outm, woutm_t, DI, Cc, Cc,          (long)ND * Cc * DI);
    cvt_t_kernel<<<nb((long)Cc * Cc, T), T, 0, stream>>>(W_out,  wout_t,  Cc, Cc, Cc,          (long)Cc * Cc);

    // 2) gather 4 directional sequences (bf16)
    gather_seq_kernel<<<nb((long)ND * nBL * Cc, T), T, 0, stream>>>(x, seq_bf, (int)(ND * nBL * Cc));

    const int Mt = (int)(nBL / 16);            // 392

    // 3) in-proj GEMM: (6272x192)@(192x768) per dir -> split x_in / z
    wmma_gemm_kernel<1, 4><<<nb((long)Mt * (2 * DI / 64) * ND, 8), 256, 0, stream>>>(
        seq_bf, nBL * Cc, Cc, win_t, (long)2 * DI * Cc,
        Mt, 2 * DI / 64, Cc, ND,
        x_in, nBL * DI, DI, z_buf, nullptr, nullptr, 0);

    // 4) depthwise causal conv -> bf16
    dwconv_kernel<<<nb((long)ND * nBL * DI, T), T, 0, stream>>>(x_in, dwk, conv_bf, (int)(ND * nBL * DI));

    // 5) pointwise GEMM + bias + SiLU -> x_conv (f32 + bf16)
    wmma_gemm_kernel<2, 4><<<nb((long)Mt * (DI / 64) * ND, 8), 256, 0, stream>>>(
        conv_bf, nBL * DI, DI, pwk_t, (long)DI * DI,
        Mt, DI / 64, DI, ND,
        x_conv, nBL * DI, DI, nullptr, x_conv_bf, cbias, DI);

    // 6) x_dbl GEMM: (6272x384)@(384x32)
    wmma_gemm_kernel<0, 2><<<nb((long)Mt * 1 * ND, 8), 256, 0, stream>>>(
        x_conv_bf, nBL * DI, DI, wx_t, (long)XDBL * DI,
        Mt, 1, DI, ND,
        x_dbl, nBL * XDBL, XDBL, nullptr, nullptr, nullptr, 0);

    // 7) delta projection + softplus
    delta_kernel<<<nb((long)ND * nBL * DI, T), T, 0, stream>>>(x_dbl, W_dt, b_dt, delta, (int)(ND * nBL * DI));

    // 8) selective scan (+ skip, gate) -> y_bf
    scan_kernel<<<dim3(ND * Bn), dim3(DI), 0, stream>>>(
        delta, x_conv, x_dbl, z_buf, A_log, D_par, y_bf);

    // 9) out-proj GEMM per dir: (6272x384)@(384x192)
    wmma_gemm_kernel<0, 4><<<nb((long)Mt * (Cc / 64) * ND, 8), 256, 0, stream>>>(
        y_bf, nBL * DI, DI, woutm_t, (long)Cc * DI,
        Mt, Cc / 64, DI, ND,
        outm, nBL * Cc, Cc, nullptr, nullptr, nullptr, 0);

    // 10) merge 4 directions (un-permute) -> bf16
    merge_kernel<<<nb((long)nBL * Cc, T), T, 0, stream>>>(outm, merged_bf, (int)(nBL * Cc));

    // 11) final GEMM: (6272x192)@(192x192) + b_out -> d_out (f32)
    wmma_gemm_kernel<5, 4><<<nb((long)Mt * (Cc / 64), 8), 256, 0, stream>>>(
        merged_bf, 0, Cc, wout_t, 0,
        Mt, Cc / 64, Cc, 1,
        out, 0, Cc, nullptr, nullptr, b_out, 0);
}
